// ChunkedLocalAttention_65687229825387
// MI455X (gfx1250) — matrix operand order repaired and checked by reading, not compiled, not run on hardware
//
#include <hip/hip_runtime.h>
#include <hip/hip_bf16.h>
#include <stdint.h>

// ---------------------------------------------------------------------------
// Chunked local attention for MI455X (gfx1250, wave32, WMMA).
// All matmuls in bf16 via v_wmma_f32_16x16x32_bf16 (f32 accumulate).
// GEMM staging uses GLOBAL_LOAD_ASYNC_TO_LDS_B128 (ASYNCcnt) when the
// toolchain exposes the builtin, with a VGPR-staged fallback otherwise.
// ---------------------------------------------------------------------------

typedef __attribute__((ext_vector_type(16))) __bf16 v16bf;
typedef __attribute__((ext_vector_type(8)))  __bf16 v8bf;
typedef __attribute__((ext_vector_type(8)))  float  v8f;

#define BATCH    4
#define SEQ      8192
#define NHEADS   16
#define HDIM     64
#define DMODEL   1024
#define CHUNKSZ  256
#define WINDOW   64
#define MTOT     (BATCH*SEQ)          // 32768

// ---- async global -> LDS copy (CDNA5), with safe fallback -------------------
#if defined(__has_builtin)
#if __has_builtin(__builtin_amdgcn_global_load_async_to_lds_b128)
#define HAVE_ASYNC_LDS 1
#endif
#endif
#ifndef HAVE_ASYNC_LDS
#define HAVE_ASYNC_LDS 0
#endif

// builtin expects pointers to 16-byte int vectors (per hipcc diagnostic)
typedef int b128v __attribute__((vector_size(16)));
typedef __attribute__((address_space(1))) b128v* gptr128;
typedef __attribute__((address_space(3))) b128v* lptr128;

static __device__ __forceinline__ void async_copy16(__bf16* lds_dst,
                                                    const __bf16* gsrc) {
#if HAVE_ASYNC_LDS
  __builtin_amdgcn_global_load_async_to_lds_b128(
      (gptr128)(gsrc), (lptr128)(lds_dst), 0, 0);
#else
  *(v8bf*)lds_dst = *(const v8bf*)gsrc;
#endif
}

static __device__ __forceinline__ void wait_async_copies() {
#if HAVE_ASYNC_LDS
#if __has_builtin(__builtin_amdgcn_s_wait_asynccnt)
  __builtin_amdgcn_s_wait_asynccnt(0);
#else
  asm volatile("s_wait_asynccnt 0x0" ::: "memory");
#endif
#endif
}

// ---- fragment loaders -------------------------------------------------------
// A-matrix (16x32 bf16) per-lane layout: row M = lane%16, two 8-element runs
// K = [off8, off8+8) and [off8+16, off8+24), off8 = (lane/16)*8.
static __device__ __forceinline__ v16bf frag_a16(const __bf16* base, int off8) {
  v8bf lo = *(const v8bf*)(base + off8);
  v8bf hi = *(const v8bf*)(base + off8 + 16);
  return __builtin_shufflevector(lo, hi, 0,1,2,3,4,5,6,7,8,9,10,11,12,13,14,15);
}
// B-matrix (32x16 bf16) per-lane layout: col N = lane%16, two 8-element runs
// K = [off8, off8+8) and [off8+16, off8+24), off8 = (lane/16)*8.  base points at the K-major row.
static __device__ __forceinline__ v16bf frag_b16(const __bf16* base, int off8) {
  v8bf lo = *(const v8bf*)(base + off8);
  v8bf hi = *(const v8bf*)(base + off8 + 16);
  return __builtin_shufflevector(lo, hi, 0,1,2,3,4,5,6,7,8,9,10,11,12,13,14,15);
}

// ---- elementwise conversions ------------------------------------------------
__global__ __launch_bounds__(256) void cvt_f32_bf16(const float* __restrict__ in,
                                                    __bf16* __restrict__ out, int n) {
  int i = (blockIdx.x * 256 + threadIdx.x) * 4;
  if (i + 3 < n) {
    float4 f = *(const float4*)(in + i);
    out[i + 0] = (__bf16)f.x;
    out[i + 1] = (__bf16)f.y;
    out[i + 2] = (__bf16)f.z;
    out[i + 3] = (__bf16)f.w;
  }
}

// wt[n*1024 + k] = (bf16) w[k*1024 + n]   (store W transposed, K-major rows)
__global__ __launch_bounds__(256) void transpose_cvt(const float* __restrict__ w,
                                                     __bf16* __restrict__ wt) {
  int idx  = blockIdx.x * 256 + threadIdx.x;   // 0 .. 1024*1024-1
  int nrow = idx >> 10;
  int k    = idx & 1023;
  wt[idx]  = (__bf16)w[k * 1024 + nrow];
}

// ---- tiled bf16 GEMM:  C[M,N] = A[M,K] * Bt[N,K]^T --------------------------
// Double-buffered LDS; async copies for tile k+1 overlap WMMAs on tile k.
template <bool OUT_BF16>
__global__ __launch_bounds__(256) void gemm_bt(const __bf16* __restrict__ A,
                                               const __bf16* __restrict__ Bt,
                                               void* __restrict__ C,
                                               int M, int N, int K) {
  __shared__ __bf16 lsA[2][128 * 32];
  __shared__ __bf16 lsB[2][128 * 32];
  const int t    = threadIdx.x;
  const int lane = t & 31;
  const int w    = t >> 5;                 // 8 waves
  const int m0   = blockIdx.x * 128;
  const int n0   = blockIdx.y * 128;
  const int wm   = (w >> 1) * 32;          // wave tile: 32 x 64
  const int wn   = (w & 1) * 64;
  const int off8  = (lane >> 4) * 8;
  const int off16 = (lane >> 4) * 16;
  const int rsel  = lane & 15;

  // stage helper: 128x32 tile (512 x 16B chunks) -> this thread does 4 chunks
  auto stage = [&](int buf, int k0) {
#pragma unroll
    for (int c = 0; c < 2; ++c) {
      int cid = t + c * 256;
      int row = cid >> 2;
      int q   = cid & 3;
      async_copy16(&lsA[buf][row * 32 + q * 8],
                   A + (size_t)(m0 + row) * K + k0 + q * 8);
      async_copy16(&lsB[buf][row * 32 + q * 8],
                   Bt + (size_t)(n0 + row) * K + k0 + q * 8);
    }
  };

  v8f acc[2][4];
#pragma unroll
  for (int mi = 0; mi < 2; ++mi)
#pragma unroll
    for (int ni = 0; ni < 4; ++ni)
      acc[mi][ni] = (v8f){0.f,0.f,0.f,0.f,0.f,0.f,0.f,0.f};

  // prologue: fill buffer 0
  stage(0, 0);
  wait_async_copies();
  __syncthreads();

  const int nIter = K >> 5;                // K/32 steps
  for (int it = 0; it < nIter; ++it) {
    const int buf = it & 1;
    if (it + 1 < nIter) {
      stage(buf ^ 1, (it + 1) * 32);       // async fill of next buffer
      __builtin_prefetch(A + (size_t)(m0 + (t >> 1)) * K + (it + 2) * 32, 0, 0);
    }

    v16bf aF[2], bF[4];
#pragma unroll
    for (int mi = 0; mi < 2; ++mi)
      aF[mi] = frag_a16(&lsA[buf][(wm + mi * 16 + rsel) * 32], off8);
#pragma unroll
    for (int ni = 0; ni < 4; ++ni)
      bF[ni] = frag_b16(&lsB[buf][(wn + ni * 16 + rsel) * 32], off8);

#pragma unroll
    for (int mi = 0; mi < 2; ++mi)
#pragma unroll
      for (int ni = 0; ni < 4; ++ni)
        acc[mi][ni] = __builtin_amdgcn_wmma_f32_16x16x32_bf16(
            false, aF[mi], false, bF[ni], (short)0, acc[mi][ni], false, false);

    wait_async_copies();                   // next buffer resident for this wave
    __syncthreads();                       // ...and for every wave
  }

  // epilogue: C/D layout -> lane holds col rsel, VGPR r holds rows r / r+8
  const int radd = (lane >> 4) * 8;
#pragma unroll
  for (int mi = 0; mi < 2; ++mi)
#pragma unroll
    for (int ni = 0; ni < 4; ++ni)
#pragma unroll
      for (int r = 0; r < 8; ++r) {
        size_t row = (size_t)(m0 + wm + mi * 16 + r + radd);
        size_t col = (size_t)(n0 + wn + ni * 16 + rsel);
        if (OUT_BF16)
          ((__bf16*)C)[row * N + col] = (__bf16)acc[mi][ni][r];
        else
          ((float*)C)[row * N + col] = acc[mi][ni][r];
      }
}

// ---- banded local attention -------------------------------------------------
// One workgroup per (batch, chunk, head).  8 waves; wave w owns q-tiles 2w,2w+1.
// Scores only for key tiles tj in [ti-4, ti]  (65-wide causal band).
#define VT_LD 288                    // Vt row length (256 + zero pad for K=96)
#define P_LD  96                     // per-wave prob scratch: 16 rows x 96 cols

__global__ __launch_bounds__(256) void attn_kernel(const __bf16* __restrict__ Q,
                                                   const __bf16* __restrict__ Km,
                                                   const __bf16* __restrict__ V,
                                                   __bf16* __restrict__ Out) {
  extern __shared__ char smem_raw[];
  __bf16* lq  = (__bf16*)smem_raw;          // 256*64
  __bf16* lk  = lq + 256 * 64;              // 256*64
  __bf16* lvt = lk + 256 * 64;              // 64*VT_LD (transposed V)
  __bf16* lp  = lvt + 64 * VT_LD;           // 8 waves * 16*P_LD

  const int blk    = blockIdx.x;
  const int head   = blk & 15;
  const int chunkn = (blk >> 4) & 31;
  const int b      = blk >> 9;
  const size_t rowbase = (size_t)b * SEQ + (size_t)chunkn * CHUNKSZ;

  const int t    = threadIdx.x;
  const int lane = t & 31;
  const int w    = t >> 5;
  const int rsel  = lane & 15;
  const int off8  = (lane >> 4) * 8;
  const int off16 = (lane >> 4) * 16;
  const int radd  = (lane >> 4) * 8;

  // ---- stage Q, K (async, row major) and V (transposed via VGPRs) ----
  {
    const __bf16* qs = Q  + (rowbase + t) * DMODEL + head * HDIM;
    const __bf16* ks = Km + (rowbase + t) * DMODEL + head * HDIM;
    const __bf16* vs = V  + (rowbase + t) * DMODEL + head * HDIM;
    v8bf vtmp[8];
#pragma unroll
    for (int q8 = 0; q8 < 8; ++q8) {
      async_copy16(lq + t * HDIM + q8 * 8, qs + q8 * 8);
      async_copy16(lk + t * HDIM + q8 * 8, ks + q8 * 8);
      vtmp[q8] = *(const v8bf*)(vs + q8 * 8);
    }
#pragma unroll
    for (int q8 = 0; q8 < 8; ++q8)
#pragma unroll
      for (int e = 0; e < 8; ++e)
        lvt[(q8 * 8 + e) * VT_LD + t] = vtmp[q8][e];
    // zero pad Vt columns 256..287 (read by the padded K=96 prob matmul)
#pragma unroll
    for (int z = 0; z < 8; ++z) {
      int idx = t * 8 + z;                 // 0..2047
      int d   = idx >> 5;
      int c   = idx & 31;
      lvt[d * VT_LD + 256 + c] = (__bf16)0.0f;
    }
  }
  wait_async_copies();
  __syncthreads();

  const float scale = 0.125f;              // 1/sqrt(64)
  __bf16* preg = lp + w * (16 * P_LD);     // per-wave scratch

  for (int qi = 0; qi < 2; ++qi) {
    const int ti   = w * 2 + qi;           // wave-uniform
    const int tj0  = (ti - 4 < 0) ? 0 : ti - 4;
    const int tcnt = ti - tj0 + 1;         // 1..5 live key tiles

    // Q fragments for both K-steps of head_dim
    v16bf aQ[2];
#pragma unroll
    for (int kc = 0; kc < 2; ++kc)
      aQ[kc] = frag_a16(lq + (ti * 16 + rsel) * HDIM + kc * 32, off8);

    // ---- scores = Q K^T over the band ----
    v8f acc[5];
#pragma unroll
    for (int s = 0; s < 5; ++s)
      acc[s] = (v8f){0.f,0.f,0.f,0.f,0.f,0.f,0.f,0.f};
#pragma unroll
    for (int s = 0; s < 5; ++s) {
      if (s < tcnt) {                      // wave-uniform branch: EXEC stays full
        const int tj = tj0 + s;
#pragma unroll
        for (int kc = 0; kc < 2; ++kc) {
          v16bf bK = frag_b16(lk + (tj * 16 + rsel) * HDIM + kc * 32, off8);
          acc[s] = __builtin_amdgcn_wmma_f32_16x16x32_bf16(
              false, aQ[kc], false, bK, (short)0, acc[s], false, false);
        }
      }
    }

    // ---- mask + softmax (rows live in VGPR index, cols across 16 lanes) ----
#pragma unroll
    for (int r = 0; r < 8; ++r) {
      const int i = ti * 16 + r + radd;    // local row in chunk
      float m = -1e30f;
#pragma unroll
      for (int s = 0; s < 5; ++s) {
        if (s < tcnt) {
          const int j = (tj0 + s) * 16 + rsel;
          float v = acc[s][r] * scale;
          bool ok = (j <= i) && (j >= i - WINDOW);
          v = ok ? v : -1e30f;
          acc[s][r] = v;
          m = fmaxf(m, v);
        }
      }
#pragma unroll
      for (int d = 1; d < 16; d <<= 1) m = fmaxf(m, __shfl_xor(m, d, 32));
      float sum = 0.f;
#pragma unroll
      for (int s = 0; s < 5; ++s) {
        if (s < tcnt) {
          float e = __expf(acc[s][r] - m);
          acc[s][r] = e;
          sum += e;
        }
      }
#pragma unroll
      for (int d = 1; d < 16; d <<= 1) sum += __shfl_xor(sum, d, 32);
      const float inv = 1.0f / sum;

      // spill normalized probs (bf16) to per-wave LDS, zero-padded to K=96
      __bf16* prow = preg + (r + radd) * P_LD;
#pragma unroll
      for (int s = 0; s < 5; ++s) {
        float pv = (s < tcnt) ? acc[s][r] * inv : 0.f;
        prow[s * 16 + rsel] = (__bf16)pv;
      }
      prow[80 + rsel] = (__bf16)0.0f;
    }

    // ---- out = probs @ V  (K = 96, padded; V read transposed) ----
    v8f oacc[4];
#pragma unroll
    for (int nd = 0; nd < 4; ++nd)
      oacc[nd] = (v8f){0.f,0.f,0.f,0.f,0.f,0.f,0.f,0.f};
#pragma unroll
    for (int kc = 0; kc < 3; ++kc) {
      v16bf aP = frag_a16(preg + rsel * P_LD + kc * 32, off8);
      const int jb = tj0 * 16 + kc * 32;
#pragma unroll
      for (int nd = 0; nd < 4; ++nd) {
        v16bf bV = frag_b16(lvt + (nd * 16 + rsel) * VT_LD + jb, off8);
        oacc[nd] = __builtin_amdgcn_wmma_f32_16x16x32_bf16(
            false, aP, false, bV, (short)0, oacc[nd], false, false);
      }
    }

    // ---- store attention output (bf16, feeds the Wo GEMM) ----
#pragma unroll
    for (int nd = 0; nd < 4; ++nd)
#pragma unroll
      for (int r = 0; r < 8; ++r) {
        size_t row = rowbase + (size_t)(ti * 16 + r + radd);
        int    col = head * HDIM + nd * 16 + rsel;
        Out[row * DMODEL + col] = (__bf16)oacc[nd][r];
      }
  }
}

// ---------------------------------------------------------------------------
extern "C" void kernel_launch(void* const* d_in, const int* in_sizes, int n_in,
                              void* d_out, int out_size, void* d_ws, size_t ws_size,
                              hipStream_t stream) {
  const float* x  = (const float*)d_in[0];
  const float* Wq = (const float*)d_in[1];
  const float* Wk = (const float*)d_in[2];
  const float* Wv = (const float*)d_in[3];
  const float* Wo = (const float*)d_in[4];

  char* ws = (char*)d_ws;
  const size_t NXB = (size_t)MTOT * DMODEL * sizeof(__bf16);   // 64 MB
  const size_t NWB = (size_t)DMODEL * DMODEL * sizeof(__bf16); //  2 MB
  __bf16* xb  = (__bf16*)(ws);
  __bf16* wtq = (__bf16*)(ws + NXB);
  __bf16* wtk = (__bf16*)(ws + NXB + 1 * NWB);
  __bf16* wtv = (__bf16*)(ws + NXB + 2 * NWB);
  __bf16* wto = (__bf16*)(ws + NXB + 3 * NWB);
  __bf16* qb  = (__bf16*)(ws + NXB + 4 * NWB);
  __bf16* kb  = (__bf16*)(ws + 2 * NXB + 4 * NWB);
  __bf16* vb  = (__bf16*)(ws + 3 * NXB + 4 * NWB);
  __bf16* ab  = (__bf16*)(ws + 4 * NXB + 4 * NWB);

  // 1) fp32 -> bf16 conversions
  const int nx = MTOT * DMODEL;                                 // 33.5M
  cvt_f32_bf16<<<nx / 1024, 256, 0, stream>>>(x, xb, nx);
  const int wgrid = (DMODEL * DMODEL) / 256;
  transpose_cvt<<<wgrid, 256, 0, stream>>>(Wq, wtq);
  transpose_cvt<<<wgrid, 256, 0, stream>>>(Wk, wtk);
  transpose_cvt<<<wgrid, 256, 0, stream>>>(Wv, wtv);
  transpose_cvt<<<wgrid, 256, 0, stream>>>(Wo, wto);

  // 2) Q/K/V projections (bf16 WMMA GEMMs)
  dim3 gg(MTOT / 128, DMODEL / 128);
  gemm_bt<true><<<gg, 256, 0, stream>>>(xb, wtq, qb, MTOT, DMODEL, DMODEL);
  gemm_bt<true><<<gg, 256, 0, stream>>>(xb, wtk, kb, MTOT, DMODEL, DMODEL);
  gemm_bt<true><<<gg, 256, 0, stream>>>(xb, wtv, vb, MTOT, DMODEL, DMODEL);

  // 3) banded local attention: one WG per (b, chunk, head)
  const size_t smem = (size_t)(256 * 64 * 2 + 64 * VT_LD + 8 * 16 * P_LD) * sizeof(__bf16);
  attn_kernel<<<BATCH * (SEQ / CHUNKSZ) * NHEADS, 256, smem, stream>>>(qb, kb, vb, ab);

  // 4) output projection -> fp32 d_out
  gemm_bt<false><<<gg, 256, 0, stream>>>(ab, wto, d_out, MTOT, DMODEL, DMODEL);
}
